// LSTM_Cell_2448131358889
// MI455X (gfx1250) — matrix operand order repaired and checked by reading, not compiled, not run on hardware
//
#include <hip/hip_runtime.h>
#include <hip/hip_bf16.h>

typedef __attribute__((ext_vector_type(16))) _Float16 v16h;
typedef __attribute__((ext_vector_type(8)))  float    v8f;
typedef int v4i __attribute__((vector_size(16)));

#define BDIM   4096   // batch
#define DIN    1024
#define UNITS  1024
#define K2     2048   // DIN + UNITS
#define M_TILE 128
#define U_TILE 64
#define K_STEP 32
#define KSTEPS (K2 / K_STEP)

union HF16 { uint4 q[2]; v16h v; };

// ---- CDNA5 async global->LDS path (ASYNCcnt), with portable fallback ------
#if defined(__AMDGCN__) && __has_builtin(__builtin_amdgcn_global_load_async_to_lds_b128)
#define ASYNC_LDS_OK 1
#endif

__device__ __forceinline__ void async_copy_b128(const _Float16* gsrc, _Float16* ldst) {
#ifdef ASYNC_LDS_OK
    // signature (from clang diagnostic): (v4i addrspace(1)*, v4i addrspace(3)*, imm, imm)
    __builtin_amdgcn_global_load_async_to_lds_b128(
        (__attribute__((address_space(1))) v4i*)gsrc,
        (__attribute__((address_space(3))) v4i*)ldst, 0, 0);
#else
    *(uint4*)ldst = *(const uint4*)gsrc;   // global_load + ds_store fallback
#endif
}

__device__ __forceinline__ void wait_async_all() {
#ifdef ASYNC_LDS_OK
#if __has_builtin(__builtin_amdgcn_s_wait_asynccnt)
    __builtin_amdgcn_s_wait_asynccnt(0);
#else
    asm volatile("s_wait_asynccnt 0x0" ::: "memory");
#endif
#endif
}

__device__ __forceinline__ float fast_sigmoid(float x) {
    return 1.0f / (1.0f + __expf(-x));
}
__device__ __forceinline__ float fast_tanh(float x) {
    // 1 - 2/(e^{2x}+1): correct saturation at +/-inf, no inf/inf NaN
    float e = __expf(2.0f * x);
    return 1.0f - 2.0f / (e + 1.0f);
}

// ---------------------------------------------------------------------------
// Kernel 1: Xh[b][k] (f16) = concat(hidden[b][0:1024], inputs[b][0:1024])
// ---------------------------------------------------------------------------
__global__ __launch_bounds__(256)
void pack_x_kernel(const float* __restrict__ inputs,
                   const float* __restrict__ hidden,
                   _Float16* __restrict__ Xh) {
    unsigned idx = blockIdx.x * 256u + threadIdx.x;
    if (idx >= (unsigned)(BDIM * K2)) return;
    unsigned b = idx >> 11;
    unsigned k = idx & 2047u;
    float v = (k < UNITS) ? hidden[b * UNITS + k]
                          : inputs[b * DIN + (k - UNITS)];
    Xh[idx] = (_Float16)v;
}

// ---------------------------------------------------------------------------
// Kernel 2: Wt[g*1024 + u][k] (f16) = W_g[k][u]   (LDS-tiled transpose)
// ---------------------------------------------------------------------------
__global__ __launch_bounds__(256)
void pack_wt_kernel(const float* __restrict__ Wf,
                    const float* __restrict__ Wi,
                    const float* __restrict__ Wc,
                    const float* __restrict__ Wo,
                    _Float16* __restrict__ Wt) {
    __shared__ float tile[32][33];
    const unsigned k0 = blockIdx.x * 32u;
    const unsigned u0 = blockIdx.y * 32u;
    const unsigned g  = blockIdx.z;
    const float* Wg = (g == 0) ? Wf : (g == 1) ? Wi : (g == 2) ? Wc : Wo;

    const unsigned tx = threadIdx.x;   // 0..31
    const unsigned ty = threadIdx.y;   // 0..7
#pragma unroll
    for (int i = 0; i < 4; ++i) {
        unsigned r = ty + 8u * i;
        tile[r][tx] = Wg[(k0 + r) * (unsigned)UNITS + u0 + tx];  // coalesced over u
    }
    __syncthreads();
#pragma unroll
    for (int i = 0; i < 4; ++i) {
        unsigned r = ty + 8u * i;          // local u index
        Wt[(g * (unsigned)UNITS + u0 + r) * (unsigned)K2 + k0 + tx] =
            (_Float16)tile[tx][r];
    }
}

// ---------------------------------------------------------------------------
// Kernel 3: fused 4-gate WMMA GEMM + LSTM epilogue
//   grid = (4096/128, 1024/64), block = 256 (8 waves, 4(M) x 2(N))
//   per wave: 32 rows x 32 units x 4 gates (2x2 tiles of 16x16 per gate)
//   double-buffered LDS, async global->LDS staging, one barrier per K-step
// ---------------------------------------------------------------------------
__global__ __launch_bounds__(256)
void lstm_gemm_kernel(const _Float16* __restrict__ Xh,   // [4096][2048]
                      const _Float16* __restrict__ Wt,   // [4096][2048] (n = g*1024+u)
                      const float* __restrict__ b_f,
                      const float* __restrict__ b_i,
                      const float* __restrict__ b_c,
                      const float* __restrict__ b_o,
                      const float* __restrict__ cell_in,
                      float* __restrict__ hidden_out,    // [4096][1024]
                      float* __restrict__ cell_out) {    // [4096][1024]
    __shared__ __align__(16) _Float16 Xs[2][M_TILE][K_STEP + 8];       // 2 x 10 KB
    __shared__ __align__(16) _Float16 Ws[2][4][U_TILE][K_STEP + 8];    // 2 x 20 KB

    const unsigned tid  = threadIdx.x;
    const unsigned lane = tid & 31u;
    const unsigned w    = tid >> 5;
    const unsigned wm   = w & 3u;            // 4 M sub-blocks of 32 rows
    const unsigned wn   = w >> 2;            // 2 N sub-blocks of 32 units
    const unsigned nlo  = lane & 15u;
    const unsigned hi   = lane >> 4;

    const unsigned m0 = blockIdx.x * (unsigned)M_TILE;
    const unsigned u0 = blockIdx.y * (unsigned)U_TILE;

    // Per-thread staging assignments (fixed across K-steps):
    //   X tile: 512 x 16B chunks  -> 2 per thread
    //   W tile: 1024 x 16B chunks -> 4 per thread
    unsigned xrow[2], xcc[2], wrowg[4], wcc[4];
#pragma unroll
    for (int it = 0; it < 2; ++it) {
        unsigned c = tid + 256u * it;
        xrow[it] = c >> 1;
        xcc[it]  = c & 1u;
    }
#pragma unroll
    for (int it = 0; it < 4; ++it) {
        unsigned c   = tid + 256u * it;
        unsigned g   = c >> 8;
        unsigned rem = c & 255u;
        wrowg[it] = g * (unsigned)UNITS + u0 + (rem >> 2);   // global row in Wt
        wcc[it]   = c;
    }

    auto load_tiles = [&](int buf, unsigned kk) {
#pragma unroll
        for (int it = 0; it < 2; ++it) {
            const _Float16* gp = Xh + (size_t)(m0 + xrow[it]) * K2 + kk + xcc[it] * 8u;
            async_copy_b128(gp, &Xs[buf][xrow[it]][xcc[it] * 8u]);
        }
#pragma unroll
        for (int it = 0; it < 4; ++it) {
            unsigned c   = wcc[it];
            unsigned g   = c >> 8;
            unsigned rem = c & 255u;
            unsigned row = rem >> 2;
            unsigned cc  = rem & 3u;
            const _Float16* gp = Wt + (size_t)wrowg[it] * K2 + kk + cc * 8u;
            async_copy_b128(gp, &Ws[buf][g][row][cc * 8u]);
        }
    };

    v8f acc[4][2][2];
#pragma unroll
    for (int g = 0; g < 4; ++g)
#pragma unroll
        for (int ti = 0; ti < 2; ++ti)
#pragma unroll
            for (int tj = 0; tj < 2; ++tj)
                acc[g][ti][tj] = (v8f)(0.0f);

    load_tiles(0, 0);                         // prologue: fill buffer 0

    for (int s = 0; s < KSTEPS; ++s) {
        wait_async_all();                     // my stores to buf done
        __syncthreads();                      // everyone's stores done, prior reads done
        const int buf = s & 1;
        if (s + 1 < KSTEPS)
            load_tiles(buf ^ 1, (unsigned)(s + 1) * K_STEP);   // overlap with WMMAs

        // ------- fragments per ISA VGPR layouts -------
        // A (16x32 f16): lane lo holds K [0,8)+[16,24); lane hi K [8,16)+[24,32)
        HF16 afr[2];
#pragma unroll
        for (int ti = 0; ti < 2; ++ti) {
            unsigned r = wm * 32u + ti * 16u + nlo;
            afr[ti].q[0] = *(const uint4*)&Xs[buf][r][8u * hi];
            afr[ti].q[1] = *(const uint4*)&Xs[buf][r][16u + 8u * hi];
        }
        // B (32x16 f16): lane lo holds K [0,8)+[16,24); lane hi K [8,16)+[24,32); N = lane&15
        HF16 bfr[4][2];
#pragma unroll
        for (int g = 0; g < 4; ++g)
#pragma unroll
            for (int tj = 0; tj < 2; ++tj) {
                unsigned r = wn * 32u + tj * 16u + nlo;
                bfr[g][tj].q[0] = *(const uint4*)&Ws[buf][g][r][8u * hi];
                bfr[g][tj].q[1] = *(const uint4*)&Ws[buf][g][r][16u + 8u * hi];
            }

#pragma unroll
        for (int g = 0; g < 4; ++g)
#pragma unroll
            for (int ti = 0; ti < 2; ++ti)
#pragma unroll
                for (int tj = 0; tj < 2; ++tj)
                    acc[g][ti][tj] = __builtin_amdgcn_wmma_f32_16x16x32_f16(
                        false, afr[ti].v, false, bfr[g][tj].v,
                        (short)0, acc[g][ti][tj], false, false);
    }

    // ------- fused LSTM epilogue (gates never hit memory) -------
#pragma unroll
    for (int tj = 0; tj < 2; ++tj) {
        unsigned gu = u0 + wn * 32u + tj * 16u + nlo;   // unit column
        float bf = b_f[gu], bi = b_i[gu], bc = b_c[gu], bo = b_o[gu];
#pragma unroll
        for (int ti = 0; ti < 2; ++ti) {
#pragma unroll
            for (int r = 0; r < 8; ++r) {
                unsigned gm = m0 + wm * 32u + ti * 16u + (unsigned)r + 8u * hi;
                float fg = fast_sigmoid(acc[0][ti][tj][r] + bf);
                float ig = fast_sigmoid(acc[1][ti][tj][r] + bi);
                float cc = fast_tanh   (acc[2][ti][tj][r] + bc);
                float og = fast_sigmoid(acc[3][ti][tj][r] + bo);
                float cprev = cell_in[gm * (unsigned)UNITS + gu];
                float cnew  = fg * cprev + ig * cc;
                float hnew  = og * fast_tanh(cnew);
                hidden_out[gm * (unsigned)UNITS + gu] = hnew;
                cell_out  [gm * (unsigned)UNITS + gu] = cnew;
            }
        }
    }
}

// ---------------------------------------------------------------------------
extern "C" void kernel_launch(void* const* d_in, const int* in_sizes, int n_in,
                              void* d_out, int out_size, void* d_ws, size_t ws_size,
                              hipStream_t stream) {
    const float* inputs  = (const float*)d_in[0];
    const float* hidden  = (const float*)d_in[1];
    const float* cell    = (const float*)d_in[2];
    const float* W_f     = (const float*)d_in[3];
    const float* b_f     = (const float*)d_in[4];
    const float* W_i     = (const float*)d_in[5];
    const float* b_i     = (const float*)d_in[6];
    const float* W_c     = (const float*)d_in[7];
    const float* b_c     = (const float*)d_in[8];
    const float* W_o     = (const float*)d_in[9];
    const float* b_o     = (const float*)d_in[10];

    float* hidden_out = (float*)d_out;                       // [4096*1024]
    float* cell_out   = hidden_out + (size_t)BDIM * UNITS;   // [4096*1024]

    _Float16* Xh = (_Float16*)d_ws;                          // 16 MB
    _Float16* Wt = Xh + (size_t)BDIM * K2;                   // 16 MB

    pack_x_kernel<<<(BDIM * K2) / 256, 256, 0, stream>>>(inputs, hidden, Xh);

    {
        dim3 grid(K2 / 32, UNITS / 32, 4);
        dim3 block(32, 8);
        pack_wt_kernel<<<grid, block, 0, stream>>>(W_f, W_i, W_c, W_o, Wt);
    }

    {
        dim3 grid(BDIM / M_TILE, UNITS / U_TILE);
        lstm_gemm_kernel<<<grid, 256, 0, stream>>>(Xh, Wt, b_f, b_i, b_c, b_o,
                                                   cell, hidden_out, cell_out);
    }
}
